// BiDAF_73435350827580
// MI455X (gfx1250) — matrix operand order repaired and checked by reading, not compiled, not run on hardware
//
#include <hip/hip_runtime.h>

typedef __attribute__((ext_vector_type(16))) _Float16 v16h;
typedef __attribute__((ext_vector_type(8)))  _Float16 h8;
typedef __attribute__((ext_vector_type(8)))  float    v8f;
typedef __attribute__((ext_vector_type(4)))  int      v4i;

// Model dims: V=32000 E=300 H=256 B=64 LQ=64 LC=512 ; 4H=1024, 2H=512, 8H=2048, KE=320
#define NTQ  4096LL   // B*LQ
#define NTC  32768LL  // B*LC

#if __has_builtin(__builtin_amdgcn_global_load_async_to_lds_b128) && \
    __has_builtin(__builtin_amdgcn_s_wait_asynccnt)
#define USE_ASYNC_LDS 1
typedef __attribute__((address_space(1))) v4i gv4i;   // global v4i
typedef __attribute__((address_space(3))) v4i lv4i;   // LDS v4i
#else
#define USE_ASYNC_LDS 0
#endif

__device__ __forceinline__ float sigf(float x) { return 1.0f / (1.0f + __expf(-x)); }

__device__ __forceinline__ v16h cat8(h8 lo, h8 hi) {
  return __builtin_shufflevector(lo, hi, 0,1,2,3,4,5,6,7,8,9,10,11,12,13,14,15);
}

// A fragment (16x32, f16): A row-major, base = &A[row0*lda + k0].
// Lane L<16 holds row M=L:     K = [0..7] then [16..23]
// Lane L>=16 holds row M=L-16: K = [8..15] then [24..31]
__device__ __forceinline__ v16h load_frag_a(const _Float16* base, int lda, int lane) {
  const int m    = lane & 15;
  const int koff = (lane & 16) ? 8 : 0;
  const _Float16* p = base + (long long)m * lda + koff;
  h8 lo = *(const h8*)(p);
  h8 hi = *(const h8*)(p + 16);
  return cat8(lo, hi);
}

// B fragment (32x16, f16) from an N-major matrix Bt[N][K]; base = &Bt[col0*ldk + k0].
// Lane n<16 holds col N=n, K=[0..7] then [16..23]; lane n>=16 holds col N=n-16, K=[8..15] then [24..31].
__device__ __forceinline__ v16h load_frag_b(const _Float16* base, int ldk, int lane) {
  const int n  = lane & 15;
  const int kb = (lane & 16) ? 8 : 0;
  const _Float16* p = base + (long long)n * ldk + kb;
  h8 lo = *(const h8*)(p);
  h8 hi = *(const h8*)(p + 16);
  return cat8(lo, hi);
}

// ---------------------------------------------------------------------------
// Generic batched WMMA GEMM: C[M,N] (f32) = A[M,K] (f16, row-major)
//                                         @ Bt[N,K] (f16, N-major == B^T layout)
// Block: 256 thr = 8 waves; tile 64(M) x 128(N); wave strip 16 x 64.
// Requires: M % 16 == 0, N % 64 == 0, K % 32 == 0. Batched via blockIdx.z.
// Wave id taken through readfirstlane so all tile bounds are scalar (SGPR)
// branches -> EXEC is all-ones around every WMMA (ISA requirement).
// ---------------------------------------------------------------------------
__global__ __launch_bounds__(256) void gemm_wmma(
    const _Float16* __restrict__ A, const _Float16* __restrict__ Bt,
    float* __restrict__ C, int M, int N, int K,
    long long sA, long long sB, long long sC)
{
  const int lane = threadIdx.x & 31;
  const int w    = __builtin_amdgcn_readfirstlane((int)(threadIdx.x >> 5)); // uniform
#if USE_ASYNC_LDS
  __shared__ _Float16 atile[8][512];    // per-wave private 16x32 staging tile
#endif
  const long long bz = blockIdx.z;
  A  += bz * sA;  Bt += bz * sB;  C += bz * sC;
  const int row0 = blockIdx.y * 64 + (w & 3) * 16;
  const int col0 = blockIdx.x * 128 + (w >> 2) * 64;
  if (row0 + 16 > M) return;            // scalar branch (no exec masking)
  if (col0 + 64 > N) return;            // scalar branch; waves own full strips
  v8f acc[4] = {};
  for (int k0 = 0; k0 < K; k0 += 32) {
#if USE_ASYNC_LDS
    // Stage this wave's 16x32 A tile into LDS with async b128 copies
    // (ASYNCcnt-tracked, wave-private region -> no barrier, just s_wait).
    _Float16* aw = &atile[w][0];
    const _Float16* gsrc = A + (long long)row0 * K + k0;
#pragma unroll
    for (int it = 0; it < 2; ++it) {
      const int chunk = lane + it * 32;       // 64 chunks of 16B
      const int r = chunk >> 2, o = (chunk & 3) * 8;
      __builtin_amdgcn_global_load_async_to_lds_b128(
          (gv4i*)(gsrc + (long long)r * K + o),
          (lv4i*)(aw + chunk * 8), 0, 0);
    }
    __builtin_amdgcn_s_wait_asynccnt(0);
    v16h a = load_frag_a(aw, 32, lane);       // ds_load path
#else
    v16h a = load_frag_a(A + (long long)row0 * K + k0, K, lane);
#endif
#pragma unroll
    for (int s = 0; s < 4; ++s) {
      v16h b = load_frag_b(Bt + (long long)(col0 + s * 16) * K + k0, K, lane);
      acc[s] = __builtin_amdgcn_wmma_f32_16x16x32_f16(false, a, false, b,
                                                      (short)0, acc[s], false, false);
    }
  }
  const int n  = lane & 15;
  const int mb = (lane & 16) ? 8 : 0;
#pragma unroll
  for (int s = 0; s < 4; ++s)
#pragma unroll
    for (int i = 0; i < 8; ++i)
      C[(long long)(row0 + mb + i) * N + col0 + s * 16 + n] = acc[s][i];
}

// ---------------------------------------------------------------------------
// Persistent bidirectional LSTM scan. grid.x = 2 (dir 0 fwd, 1 bwd),
// block = 1024 threads = 32 waves. h ([64,256] f16) lives in LDS, cell state
// in registers. Per step: gates = xW[t] + bias + h @ Whh^T via WMMA, where
// wave w owns batch rows [ (w&3)*16, +16 ) and hidden cols [ (w>>2)*32, +32 )
// of ALL FOUR gates -> nonlinearity runs straight out of the accumulators.
//   xW   : [2][B*T][1024]  (precomputed input projection, fp32)
//   bias : [2][1024]
//   Whh  : [2][1024][256] f16 (native N-major layout of Whh)
//   h_out: [B][T][512] fp32 (this dir writes cols dir*256 .. +255)
// ---------------------------------------------------------------------------
__global__ __launch_bounds__(1024) void lstm_scan(
    const float* __restrict__ xW, const float* __restrict__ bias,
    const _Float16* __restrict__ Whh, float* __restrict__ h_out, int T)
{
  const int dir  = blockIdx.x;
  const int tid  = threadIdx.x;
  const int lane = tid & 31;
  const int w    = __builtin_amdgcn_readfirstlane((int)(tid >> 5));  // uniform
  const int r0   = (w & 3) * 16;   // batch-row tile   (SGPR)
  const int j0   = (w >> 2) * 32;  // hidden-col group (SGPR)

  __shared__ _Float16 h_lds[64][256];
  {
    _Float16* p = &h_lds[0][0];
#pragma unroll
    for (int i = 0; i < 16; ++i) p[tid * 16 + i] = (_Float16)0.f;
  }

  const int n  = lane & 15;
  const int mb = (lane & 16) ? 8 : 0;

  float bI[2], bF[2], bG[2], bO[2];
#pragma unroll
  for (int s = 0; s < 2; ++s) {
    const int jc = j0 + s * 16 + n;
    bI[s] = bias[dir * 1024 + jc];
    bF[s] = bias[dir * 1024 + 256 + jc];
    bG[s] = bias[dir * 1024 + 512 + jc];
    bO[s] = bias[dir * 1024 + 768 + jc];
  }
  float cstate[16];
#pragma unroll
  for (int i = 0; i < 16; ++i) cstate[i] = 0.f;

  const _Float16* Wd  = Whh + (long long)dir * 1024 * 256;
  const float*    xWd = xW  + (long long)dir * 64 * T * 1024;
  __syncthreads();

  for (int t = 0; t < T; ++t) {
    const int tt = dir ? (T - 1 - t) : t;
    v8f acc[8] = {};
    for (int k0 = 0; k0 < 256; k0 += 32) {
      v16h a = load_frag_a(&h_lds[r0][k0], 256, lane);   // ds_load path
#pragma unroll
      for (int g = 0; g < 4; ++g) {
#pragma unroll
        for (int s = 0; s < 2; ++s) {
          v16h b = load_frag_b(Wd + (long long)(g * 256 + j0 + s * 16) * 256 + k0,
                               256, lane);
          acc[g * 2 + s] = __builtin_amdgcn_wmma_f32_16x16x32_f16(
              false, a, false, b, (short)0, acc[g * 2 + s], false, false);
        }
      }
    }
    __syncthreads();   // all waves done reading h_lds before it is rewritten

#pragma unroll
    for (int s = 0; s < 2; ++s) {
#pragma unroll
      for (int i = 0; i < 8; ++i) {
        const int brow = r0 + mb + i;
        const int j    = j0 + s * 16 + n;
        const float* xr = xWd + ((long long)brow * T + tt) * 1024;
        const float gi = acc[0 + s][i] + xr[j      ] + bI[s];
        const float gf = acc[2 + s][i] + xr[j + 256] + bF[s];
        const float gg = acc[4 + s][i] + xr[j + 512] + bG[s];
        const float go = acc[6 + s][i] + xr[j + 768] + bO[s];
        float cc = sigf(gf) * cstate[s * 8 + i] + sigf(gi) * tanhf(gg);
        const float hh = sigf(go) * tanhf(cc);
        cstate[s * 8 + i] = cc;
        h_lds[brow][j] = (_Float16)hh;
        h_out[((long long)brow * T + tt) * 512 + dir * 256 + j] = hh;
      }
    }
    if (t + 1 < T) {   // gfx1250 global_prefetch_b8 on next step's xW rows
      const int tn = dir ? (T - 2 - t) : (t + 1);
      __builtin_prefetch(xWd + ((long long)(r0 + mb) * T + tn) * 1024 + j0, 0, 0);
    }
    __syncthreads();
  }
}

// ---------------------------------------------------------------------------
// Small helper kernels
// ---------------------------------------------------------------------------
__global__ void embed_f16_kernel(const int* __restrict__ tok, const float* __restrict__ emb,
                                 _Float16* __restrict__ out, long long total, int KE) {
  const long long i = (long long)blockIdx.x * blockDim.x + threadIdx.x;
  if (i >= total) return;
  const int k = (int)(i % KE);
  const long long r = i / KE;
  float v = 0.f;
  if (k < 300) v = emb[(long long)tok[r] * 300 + k];
  out[i] = (_Float16)v;
}

__global__ void cast_pad_kernel(const float* __restrict__ src, _Float16* __restrict__ dst,
                                long long rows, int cols, int colsPad) {
  const long long i = (long long)blockIdx.x * blockDim.x + threadIdx.x;
  if (i >= rows * colsPad) return;
  const int k = (int)(i % colsPad);
  const long long r = i / colsPad;
  dst[i] = (_Float16)((k < cols) ? src[r * cols + k] : 0.f);
}

__global__ void transpose_qT_kernel(const _Float16* __restrict__ in, _Float16* __restrict__ out) {
  // [64][64][512] -> [64][512][64]
  const long long i = (long long)blockIdx.x * blockDim.x + threadIdx.x;
  if (i >= 64LL * 64 * 512) return;
  const int cc = (int)(i % 512);
  const long long rem = i / 512;
  const int rr = (int)(rem % 64);
  const int bb = (int)(rem / 64);
  out[((long long)bb * 512 + cc) * 64 + rr] = in[i];
}

__global__ void scale_wcq_kernel(const float* __restrict__ cOut, const float* __restrict__ simw,
                                 _Float16* __restrict__ o, long long total) {
  const long long i = (long long)blockIdx.x * blockDim.x + threadIdx.x;
  if (i >= total) return;
  const int h = (int)(i % 512);
  o[i] = (_Float16)(cOut[i] * simw[1024 + h]);
}

__global__ void rowdot_kernel(const float* __restrict__ X, const float* __restrict__ wvec,
                              float* __restrict__ out, int rows) {
  const int r = blockIdx.x * blockDim.x + threadIdx.x;
  if (r >= rows) return;
  float s = 0.f;
  for (int h = 0; h < 512; ++h) s += X[(long long)r * 512 + h] * wvec[h];
  out[r] = s;
}

// per (b,c) row: add bias terms, softmax over q (64), emit f16 probs + row max
__global__ __launch_bounds__(64) void softmax_q_kernel(
    const float* __restrict__ S, const float* __restrict__ rowc,
    const float* __restrict__ colq, const float* __restrict__ simb,
    _Float16* __restrict__ aOut, float* __restrict__ Smax) {
  const int r = blockIdx.x, t = threadIdx.x;
  const int b = r >> 9;
  const float v = S[(long long)r * 64 + t] + rowc[r] + colq[b * 64 + t] + simb[0];
  __shared__ float red[64];
  red[t] = v; __syncthreads();
  for (int o = 32; o > 0; o >>= 1) { if (t < o) red[t] = fmaxf(red[t], red[t + o]); __syncthreads(); }
  const float m = red[0]; __syncthreads();
  if (t == 0) Smax[r] = m;
  const float e = __expf(v - m);
  red[t] = e; __syncthreads();
  for (int o = 32; o > 0; o >>= 1) { if (t < o) red[t] += red[t + o]; __syncthreads(); }
  aOut[(long long)r * 64 + t] = (_Float16)(e / red[0]);
}

__global__ __launch_bounds__(512) void batt_kernel(const float* __restrict__ Smax,
                                                   float* __restrict__ batt) {
  const int b = blockIdx.x, t = threadIdx.x;
  __shared__ float red[512];
  const float v = Smax[b * 512 + t];
  red[t] = v; __syncthreads();
  for (int o = 256; o > 0; o >>= 1) { if (t < o) red[t] = fmaxf(red[t], red[t + o]); __syncthreads(); }
  const float m = red[0]; __syncthreads();
  const float e = __expf(v - m);
  red[t] = e; __syncthreads();
  for (int o = 256; o > 0; o >>= 1) { if (t < o) red[t] += red[t + o]; __syncthreads(); }
  batt[b * 512 + t] = e / red[0];
}

__global__ void q2c_kernel(const float* __restrict__ batt, const float* __restrict__ cOut,
                           float* __restrict__ q2c) {
  const int b = blockIdx.x;
  const int h = blockIdx.y * 128 + threadIdx.x;
  float s = 0.f;
  for (int cpos = 0; cpos < 512; ++cpos)
    s += batt[b * 512 + cpos] * cOut[((long long)(b * 512 + cpos)) * 512 + h];
  q2c[b * 512 + h] = s;
}

__global__ void buildG_kernel(const float* __restrict__ cOut, const float* __restrict__ c2q,
                              const float* __restrict__ q2c, _Float16* __restrict__ G) {
  const long long i = (long long)blockIdx.x * blockDim.x + threadIdx.x;
  if (i >= NTC * 512) return;
  const int h = (int)(i % 512);
  const long long row = i / 512;
  const int b = (int)(row >> 9);
  const float co = cOut[i], cq = c2q[i], qc = q2c[b * 512 + h];
  _Float16* g = G + row * 2048;
  g[h]        = (_Float16)co;
  g[512 + h]  = (_Float16)cq;
  g[1024 + h] = (_Float16)(co * cq);
  g[1536 + h] = (_Float16)(co * qc);
}

__global__ void startend_kernel(const float* __restrict__ Mo, const float* __restrict__ sw,
                                const float* __restrict__ sb, const float* __restrict__ ew,
                                const float* __restrict__ eb, float* __restrict__ out) {
  const int r = blockIdx.x * blockDim.x + threadIdx.x;
  if (r >= 32768) return;
  float s = 0.f, e = 0.f;
  for (int h = 0; h < 512; ++h) {
    const float m = Mo[(long long)r * 512 + h];
    s += m * sw[h];
    e += m * ew[h];
  }
  out[r] = s + sb[0];
  out[32768 + r] = e + eb[0];
}

// ---------------------------------------------------------------------------
// Host orchestration
// ---------------------------------------------------------------------------
extern "C" void kernel_launch(void* const* d_in, const int* in_sizes, int n_in,
                              void* d_out, int out_size, void* d_ws, size_t ws_size,
                              hipStream_t stream) {
  (void)in_sizes; (void)n_in; (void)out_size; (void)ws_size;
  const int*   q       = (const int*)d_in[0];
  const int*   c       = (const int*)d_in[1];
  const float* emb     = (const float*)d_in[2];
  const float* Wih_q   = (const float*)d_in[3];
  const float* Whh_q   = (const float*)d_in[4];
  const float* b_q     = (const float*)d_in[5];
  const float* Wih_c   = (const float*)d_in[6];
  const float* Whh_c   = (const float*)d_in[7];
  const float* b_c     = (const float*)d_in[8];
  const float* Wih_m   = (const float*)d_in[9];
  const float* Whh_m   = (const float*)d_in[10];
  const float* b_m     = (const float*)d_in[11];
  const float* sim_w   = (const float*)d_in[12];
  const float* sim_b   = (const float*)d_in[13];
  const float* start_w = (const float*)d_in[14];
  const float* start_b = (const float*)d_in[15];
  const float* end_w   = (const float*)d_in[16];
  const float* end_b   = (const float*)d_in[17];
  float* out = (float*)d_out;

  char* ws = (char*)d_ws;
  size_t off = 0;
  auto alloc = [&](size_t bytes) -> void* {
    void* p = ws + off;
    off = (off + bytes + 255) & ~(size_t)255;
    return p;
  };
  auto cdiv = [](long long a, long long b) -> unsigned { return (unsigned)((a + b - 1) / b); };

  _Float16* qe    = (_Float16*)alloc(NTQ * 320 * 2);
  _Float16* ce    = (_Float16*)alloc(NTC * 320 * 2);
  _Float16* wihq  = (_Float16*)alloc(2LL * 1024 * 320 * 2);
  _Float16* wihc  = (_Float16*)alloc(2LL * 1024 * 320 * 2);
  _Float16* wihm  = (_Float16*)alloc(2LL * 1024 * 2048 * 2);
  _Float16* whhq  = (_Float16*)alloc(2LL * 1024 * 256 * 2);
  _Float16* whhc  = (_Float16*)alloc(2LL * 1024 * 256 * 2);
  _Float16* whhm  = (_Float16*)alloc(2LL * 1024 * 256 * 2);
  float*    xwq   = (float*)alloc(2LL * NTQ * 1024 * 4);
  float*    xwc   = (float*)alloc(2LL * NTC * 1024 * 4);   // reused for xW_m
  float*    qout  = (float*)alloc(NTQ * 512 * 4);
  float*    cout_ = (float*)alloc(NTC * 512 * 4);
  _Float16* qf16  = (_Float16*)alloc(NTQ * 512 * 2);
  _Float16* qT    = (_Float16*)alloc(NTQ * 512 * 2);
  _Float16* cwcq  = (_Float16*)alloc(NTC * 512 * 2);
  float*    rowc  = (float*)alloc(NTC * 4);
  float*    colq  = (float*)alloc(NTQ * 4);
  float*    Sbuf  = (float*)alloc(NTC * 64 * 4);
  float*    Smax  = (float*)alloc(NTC * 4);
  _Float16* aF    = (_Float16*)alloc(NTC * 64 * 2);
  float*    battp = (float*)alloc(64LL * 512 * 4);
  float*    c2q   = (float*)alloc(NTC * 512 * 4);
  float*    q2cp  = (float*)alloc(64LL * 512 * 4);
  _Float16* Gb    = (_Float16*)alloc(NTC * 2048 * 2);
  float*    mout  = (float*)alloc(NTC * 512 * 4);

  // 1) embeddings -> f16, K padded 300 -> 320
  embed_f16_kernel<<<cdiv(NTQ * 320, 256), 256, 0, stream>>>(q, emb, qe, NTQ * 320, 320);
  embed_f16_kernel<<<cdiv(NTC * 320, 256), 256, 0, stream>>>(c, emb, ce, NTC * 320, 320);

  // 2) weights -> f16 (native N-major layout is already B^T for our GEMM)
  cast_pad_kernel<<<cdiv(2048LL * 320, 256), 256, 0, stream>>>(Wih_q, wihq, 2048, 300, 320);
  cast_pad_kernel<<<cdiv(2048LL * 320, 256), 256, 0, stream>>>(Wih_c, wihc, 2048, 300, 320);
  cast_pad_kernel<<<cdiv(2048LL * 2048, 256), 256, 0, stream>>>(Wih_m, wihm, 2048, 2048, 2048);
  cast_pad_kernel<<<cdiv(2048LL * 256, 256), 256, 0, stream>>>(Whh_q, whhq, 2048, 256, 256);
  cast_pad_kernel<<<cdiv(2048LL * 256, 256), 256, 0, stream>>>(Whh_c, whhc, 2048, 256, 256);
  cast_pad_kernel<<<cdiv(2048LL * 256, 256), 256, 0, stream>>>(Whh_m, whhm, 2048, 256, 256);

  // 3) input projections xW = x @ Wih^T (batched over the 2 directions)
  gemm_wmma<<<dim3(8, 64, 2), 256, 0, stream>>>(qe, wihq, xwq, 4096, 1024, 320,
                                                0, 1024LL * 320, NTQ * 1024);
  gemm_wmma<<<dim3(8, 512, 2), 256, 0, stream>>>(ce, wihc, xwc, 32768, 1024, 320,
                                                 0, 1024LL * 320, NTC * 1024);

  // 4) bidirectional recurrences (persistent workgroups, WMMA per step)
  lstm_scan<<<2, 1024, 0, stream>>>(xwq, b_q, whhq, qout, 64);
  lstm_scan<<<2, 1024, 0, stream>>>(xwc, b_c, whhc, cout_, 512);

  // 5) attention prep
  cast_pad_kernel<<<cdiv(NTQ * 512, 256), 256, 0, stream>>>(qout, qf16, NTQ, 512, 512);
  transpose_qT_kernel<<<cdiv(NTQ * 512, 256), 256, 0, stream>>>(qf16, qT);
  scale_wcq_kernel<<<cdiv(NTC * 512, 256), 256, 0, stream>>>(cout_, sim_w, cwcq, NTC * 512);
  rowdot_kernel<<<cdiv(NTC, 256), 256, 0, stream>>>(cout_, sim_w, rowc, 32768);
  rowdot_kernel<<<cdiv(NTQ, 256), 256, 0, stream>>>(qout, sim_w + 512, colq, 4096);

  // 6) S[b] = (c_out*wcq) @ q_out^T  : M=512, N=64, K=512, batch=64
  gemm_wmma<<<dim3(1, 8, 64), 256, 0, stream>>>(cwcq, qf16, Sbuf, 512, 64, 512,
                                                512LL * 512, 64LL * 512, 512LL * 64);
  softmax_q_kernel<<<32768, 64, 0, stream>>>(Sbuf, rowc, colq, sim_b, aF, Smax);
  batt_kernel<<<64, 512, 0, stream>>>(Smax, battp);
  q2c_kernel<<<dim3(64, 4), 128, 0, stream>>>(battp, cout_, q2cp);

  // 7) c2q[b] = a[b] @ q_out[b] : M=512, N=512, K=64, batch=64
  gemm_wmma<<<dim3(4, 8, 64), 256, 0, stream>>>(aF, qT, c2q, 512, 512, 64,
                                                512LL * 64, 512LL * 64, 512LL * 512);

  // 8) G = [c_out, c2q, c_out*c2q, c_out*q2c] in f16
  buildG_kernel<<<cdiv(NTC * 512, 256), 256, 0, stream>>>(cout_, c2q, q2cp, Gb);

  // 9) modeling LSTM: xW_m = G @ Wih_m^T (reuse xwc), then scan
  gemm_wmma<<<dim3(8, 512, 2), 256, 0, stream>>>(Gb, wihm, xwc, 32768, 1024, 2048,
                                                 0, 1024LL * 2048, NTC * 1024);
  lstm_scan<<<2, 1024, 0, stream>>>(xwc, b_m, whhm, mout, 512);

  // 10) heads -> d_out = [start(32768) | end(32768)]
  startend_kernel<<<cdiv(32768, 256), 256, 0, stream>>>(mout, start_w, start_b,
                                                        end_w, end_b, out);
}
